// MoSRAHLayer_67190468378644
// MI455X (gfx1250) — matrix operand order repaired and checked by reading, not compiled, not run on hardware
//
#include <hip/hip_runtime.h>

// MoE routed-attention for MI455X (gfx1250, wave32).
// - All GEMMs via v_wmma_f32_16x16x32_bf16 (compute-bound: ~103 GFLOP vs ~110MB
//   of traffic that fits in the 192MB L2).
// - Attention K/V tiles staged into LDS by the Tensor Data Mover
//   (tensor_load_to_lds, double-buffered on TENSORcnt) with TDM padding chosen
//   for LDS bank spread; fragments read back with ds_load_b128.
// - Query index kept in the lane/column dimension through S^T = K*Q^T so the
//   C-layout -> B-fragment conversion between chained WMMAs needs only a
//   shfl_xor(16) per register, no LDS round-trip.

#define DEV __device__ __forceinline__

constexpr int Bq = 8;
constexpr int Nq = 1024;
constexpr int Dq = 1024;
constexpr int Eq = 8;
constexpr int Kq = 2;
constexpr int Hq = 128;
constexpr int BN = Bq * Nq;

typedef __attribute__((ext_vector_type(16))) __bf16 v16bf;
typedef __attribute__((ext_vector_type(8)))  float  v8f;
typedef __attribute__((ext_vector_type(4)))  unsigned int u32x4;
typedef __attribute__((ext_vector_type(8)))  int      i32x8;
typedef __attribute__((ext_vector_type(4)))  int      i32x4;

union AB16 { v16bf v; uint4 q[2]; unsigned short s[16]; };
union PK8  { unsigned short s[8]; uint4 q; };

DEV unsigned short f2bfbits(float f) {
  unsigned u = __float_as_uint(f);
  unsigned r = u + 0x7FFFu + ((u >> 16) & 1u);   // round-to-nearest-even
  return (unsigned short)(r >> 16);
}
DEV void stbf(__bf16* p, float f) { *reinterpret_cast<unsigned short*>(p) = f2bfbits(f); }
DEV uint4 ld16(const __bf16* p) { return *reinterpret_cast<const uint4*>(p); }

DEV v8f wmma_bf16(const AB16& a, const AB16& b, v8f c) {
  return __builtin_amdgcn_wmma_f32_16x16x32_bf16(false, a.v, false, b.v, (short)0, c,
                                                 false, false);
}

// Two stacked C-layout 16-row tiles along K  ->  B-matrix bf16 fragment.
// B order built here: lane col = L%16; K = 0..15 (lanes 0-15) / 16..31 (lanes 16-31).
// C layout: lane col = L%16; rows = (L>=16)*8 + r.  One shfl_xor(16) per reg.
DEV AB16 bfrag(const v8f& t0, const v8f& t1, int halfid) {
  AB16 r;
#pragma unroll
  for (int i = 0; i < 8; ++i) {
    float o0 = __shfl_xor(t0[i], 16, 32);
    float o1 = __shfl_xor(t1[i], 16, 32);
    r.s[i]     = f2bfbits(halfid ? o1    : t0[i]);
    r.s[8 + i] = f2bfbits(halfid ? t1[i] : o0);
  }
  return r;
}

// ------------------------------------------------------------ TDM descriptor
// 2D tile load, bf16 elements (data_size=2B). D# layout per CDNA5 ISA 8.3/8.4.
// pad_interval_code: pad after (1<<code)*8 bytes; pad_amount_code: (code+1) dwords.
DEV void tdm_load_2d(unsigned lds_off, const void* gaddr,
                     unsigned tile_d0, unsigned tile_d1,
                     unsigned tensor_d0, unsigned tensor_d1,
                     unsigned long long stride0_elems,
                     unsigned pad_interval_code, unsigned pad_amount_code) {
  unsigned long long ga = (unsigned long long)(uintptr_t)gaddr;
  u32x4 g0;
  g0[0] = 1u;                                         // count=1, user descriptor
  g0[1] = lds_off;                                    // lds_addr (bytes)
  g0[2] = (unsigned)(ga & 0xFFFFFFFFu);               // global_addr[31:0]
  g0[3] = (unsigned)((ga >> 32) & 0x01FFFFFFu)        // global_addr[56:32]
          | (2u << 30);                               // type = 2 ("image")
  i32x8 g1;
  unsigned dw0 = (1u << 16);                          // data_size = 1 -> 2 bytes
  if (pad_amount_code | pad_interval_code)
    dw0 |= (1u << 20) | (pad_interval_code << 22) | (pad_amount_code << 25);
  g1[0] = (int)dw0;
  g1[1] = (int)((tensor_d0 & 0xFFFFu) << 16);                       // dim0[15:0]
  g1[2] = (int)(((tensor_d0 >> 16) & 0xFFFFu) | ((tensor_d1 & 0xFFFFu) << 16));
  g1[3] = (int)(((tensor_d1 >> 16) & 0xFFFFu) | ((tile_d0 & 0xFFFFu) << 16));
  g1[4] = (int)(tile_d1 & 0xFFFFu);                                 // tile_dim2=0
  g1[5] = (int)(unsigned)(stride0_elems & 0xFFFFFFFFull);
  g1[6] = (int)(unsigned)((stride0_elems >> 32) & 0xFFFFull);       // dim1_stride=0
  g1[7] = 0;
  i32x4 z4 = {0, 0, 0, 0};
#if __has_include(<hip/amd_detail/amd_gfx1250_TDM.h>)
  i32x8 z8 = {0, 0, 0, 0, 0, 0, 0, 0};
  __builtin_amdgcn_tensor_load_to_lds(g0, g1, z4, z4, z8, 0);
#else
  __builtin_amdgcn_tensor_load_to_lds(g0, g1, z4, z4, 0);
#endif
}

// ---------------------------------------------------------------- converts
__global__ void k_cvt(const float* __restrict__ src, __bf16* __restrict__ dst, int n) {
  int i = blockIdx.x * blockDim.x + threadIdx.x;
  if (i < n) stbf(dst + i, src[i]);
}

// src (E,R,C) f32  ->  dst (E,C,R) bf16
__global__ void k_tr(const float* __restrict__ src, __bf16* __restrict__ dst, int R, int C) {
  int i = blockIdx.x * blockDim.x + threadIdx.x;
  if (i >= Eq * R * C) return;
  int cc = i % C, rr = (i / C) % R, e = i / (R * C);
  stbf(dst + ((size_t)e * C + cc) * R + rr, src[i]);
}

// ---------------------------------------------------------------- router
__global__ __launch_bounds__(32)
void k_router(const float* __restrict__ x, const float* __restrict__ wr,
              float* __restrict__ rw, float* __restrict__ counts,
              float* __restrict__ probsum) {
  int m = blockIdx.x, lane = threadIdx.x;
  float acc[Eq];
#pragma unroll
  for (int e = 0; e < Eq; ++e) acc[e] = 0.f;
  const float* xr = x + (size_t)m * Dq;
  for (int d = lane; d < Dq; d += 32) {
    float xv = xr[d];
    const float* w = wr + (size_t)d * Eq;
#pragma unroll
    for (int e = 0; e < Eq; ++e) acc[e] += xv * w[e];
  }
#pragma unroll
  for (int e = 0; e < Eq; ++e)
    for (int off = 16; off; off >>= 1) acc[e] += __shfl_xor(acc[e], off, 32);
  if (lane == 0) {
    float mx = acc[0];
#pragma unroll
    for (int e = 1; e < Eq; ++e) mx = fmaxf(mx, acc[e]);
    float p[Eq], s = 0.f;
#pragma unroll
    for (int e = 0; e < Eq; ++e) { p[e] = __expf(acc[e] - mx); s += p[e]; }
#pragma unroll
    for (int e = 0; e < Eq; ++e) p[e] /= s;
    int i1 = 0;
#pragma unroll
    for (int e = 1; e < Eq; ++e) if (p[e] > p[i1]) i1 = e;
    int i2 = (i1 == 0) ? 1 : 0;
#pragma unroll
    for (int e = 0; e < Eq; ++e) if (e != i1 && p[e] > p[i2]) i2 = e;
    float tot = p[i1] + p[i2];
    rw[(size_t)i1 * BN + m] = p[i1] / tot;
    rw[(size_t)i2 * BN + m] = p[i2] / tot;
    atomicAdd(&counts[i1], 1.f);
    atomicAdd(&counts[i2], 1.f);
#pragma unroll
    for (int e = 0; e < Eq; ++e) atomicAdd(&probsum[e], p[e]);
  }
}

// ---------------------------------------------------------------- QKV proj + RoPE
// One wave per 16-token x 128(H) tile; 8 column WMMA tiles; RoPE in-register.
__global__ __launch_bounds__(32)
void k_qkv(const __bf16* __restrict__ xbf, const __bf16* __restrict__ wtq,
           const __bf16* __restrict__ wtk, const __bf16* __restrict__ wtv,
           const int* __restrict__ pids, __bf16* __restrict__ qbf,
           __bf16* __restrict__ kbf, __bf16* __restrict__ vt) {
  int mt = blockIdx.x, e = blockIdx.y, proj = blockIdx.z;
  int lane = threadIdx.x, c = lane & 15, halfid = lane >> 4;
  const __bf16* wt = (proj == 0 ? wtq : proj == 1 ? wtk : wtv) + (size_t)e * Hq * Dq;
  int m0 = mt * 16;

  v8f acc[8];
#pragma unroll
  for (int t = 0; t < 8; ++t)
#pragma unroll
    for (int r = 0; r < 8; ++r) acc[t][r] = 0.f;

  const __bf16* xrow = xbf + (size_t)(m0 + c) * Dq;   // A row = token m0+c
  for (int kc = 0; kc < Dq; kc += 32) {
    __builtin_prefetch(xrow + kc + 256, 0, 1);        // global_prefetch_b8
    AB16 a;                                           // A: K = {16*halfid..16*halfid+15}
    a.q[0] = ld16(xrow + kc + halfid * 16);
    a.q[1] = ld16(xrow + kc + halfid * 16 + 8);
#pragma unroll
    for (int t = 0; t < 8; ++t) {
      const __bf16* wrow = wt + (size_t)(t * 16 + c) * Dq + kc + halfid * 16;
      AB16 b; b.q[0] = ld16(wrow); b.q[1] = ld16(wrow + 8);
      acc[t] = wmma_bf16(a, b, acc[t]);
    }
  }

  if (proj == 2) {                                    // V: store transposed [E][H][M]
#pragma unroll
    for (int t = 0; t < 8; ++t) {
      PK8 pk;
#pragma unroll
      for (int r = 0; r < 8; ++r) pk.s[r] = f2bfbits(acc[t][r]);
      *reinterpret_cast<uint4*>(vt + ((size_t)e * Hq + t * 16 + c) * BN +
                                m0 + halfid * 8) = pk.q;
    }
  } else {                                            // Q/K: RoPE then store [E][M][H]
    __bf16* dst = (proj == 0 ? qbf : kbf) + (size_t)e * BN * Hq;
#pragma unroll
    for (int t = 0; t < 4; ++t) {
      int j = t * 16 + c;                             // j in 0..63, pairs with j+64
      float invf = __expf(-(float)j * 0.14391156516f);  // ln(1e4)/64
#pragma unroll
      for (int r = 0; r < 8; ++r) {
        int m = m0 + halfid * 8 + r;
        float sn, cs;
        __sincosf((float)pids[m] * invf, &sn, &cs);
        float x1 = acc[t][r], x2 = acc[t + 4][r];
        __bf16* o = dst + (size_t)m * Hq;
        stbf(o + j,      x1 * cs - x2 * sn);
        stbf(o + j + 64, x1 * sn + x2 * cs);
      }
    }
  }
}

// ---------------------------------------------------------------- attention + out-proj
// LDS staging geometry (uint4 = 16B units). TDM pads 4 dwords per row:
//  K tile: 32 rows x 256B  -> 272B pitch (17 u4)   pad_interval=5 (256B), amount=3
//  V tile: 128 rows x 64B  -> 80B  pitch (5 u4)    pad_interval=3 (64B),  amount=3
constexpr int KROW16  = 17;
constexpr int KTILE16 = 32 * KROW16;    // 544
constexpr int VROW16  = 5;
constexpr int VTILE16 = 128 * VROW16;   // 640
constexpr int BUF16   = KTILE16 + VTILE16;  // 1184 u4 = 18944 B per buffer

__global__ __launch_bounds__(32)
void k_attn(const __bf16* __restrict__ qbf, const __bf16* __restrict__ kbf,
            const __bf16* __restrict__ vt, const __bf16* __restrict__ wotd,
            const float* __restrict__ rw, const int* __restrict__ pids,
            float* __restrict__ out) {
  __shared__ uint4 shmem[2][BUF16];     // 37.9 KB of the 320 KB/WGP

  int qt = blockIdx.x, b = blockIdx.y, e = blockIdx.z;
  int lane = threadIdx.x, c = lane & 15, halfid = lane >> 4;
  int n0 = qt * 16;
  int qtok = b * Nq + n0 + c;

  float rwq = rw[(size_t)e * BN + qtok];
  bool actq = rwq > 0.f;
  if (__ballot(actq) == 0ull) return;                 // uniform: no token picked e
  int posq = pids[qtok];

  AB16 qf[4];                                         // Q^T B-fragments (persistent)
  const __bf16* qrow = qbf + ((size_t)e * BN + qtok) * Hq;
#pragma unroll
  for (int ch = 0; ch < 4; ++ch) {
    qf[ch].q[0] = ld16(qrow + ch * 32 + halfid * 8);
    qf[ch].q[1] = ld16(qrow + ch * 32 + 16 + halfid * 8);
  }

  v8f oacc[8];
#pragma unroll
  for (int t = 0; t < 8; ++t)
#pragma unroll
    for (int r = 0; r < 8; ++r) oacc[t][r] = 0.f;
  float mrun = -3.0e38f, lrun = 0.f;
  const float scale = 0.0883883476f;                  // 1/sqrt(128)

  // ---- TDM staging: K tile + V^T tile for 32 keys into LDS buffer `buf`
  auto stage = [&](int kt, int buf) {
    int kbase = kt * 32;
    unsigned lds0 = (unsigned)(uintptr_t)(&shmem[buf][0]);
    const __bf16* kg = kbf + ((size_t)e * BN + b * Nq + kbase) * Hq;
    tdm_load_2d(lds0, kg, /*tile*/128, 32, /*tensor*/128, 32,
                /*stride0*/Hq, /*pad*/5, 3);
    const __bf16* vg = vt + (size_t)e * Hq * BN + b * Nq + kbase;
    tdm_load_2d(lds0 + KTILE16 * 16, vg, /*tile*/32, 128, /*tensor*/32, 128,
                /*stride0*/BN, /*pad*/3, 3);
  };

  int ktmax = (n0 + 15) / 32;                         // causal upper bound (uniform)
  stage(0, 0);
  for (int kt = 0; kt <= ktmax; ++kt) {
    int buf = kt & 1;
    if (kt < ktmax) {                                 // prefetch next, keep 2 in flight
      stage(kt + 1, buf ^ 1);
      __builtin_amdgcn_s_wait_tensorcnt(2);           // current buffer complete
    } else {
      __builtin_amdgcn_s_wait_tensorcnt(0);
    }
    int kbase = kt * 32;

    v8f st[2];
#pragma unroll
    for (int s = 0; s < 2; ++s) {                     // S^T = K * Q^T (16 keys/subtile)
      int ms0 = kbase + s * 16;
      int krow = (s * 16 + c) * KROW16;               // LDS row base (u4 units)
      v8f sacc;
#pragma unroll
      for (int r = 0; r < 8; ++r) sacc[r] = 0.f;
#pragma unroll
      for (int ch = 0; ch < 4; ++ch) {
        AB16 kf;                                      // ds_load_b128 x2
        kf.q[0] = shmem[buf][krow + ch * 4 + halfid];
        kf.q[1] = shmem[buf][krow + ch * 4 + 2 + halfid];
        sacc = wmma_bf16(kf, qf[ch], sacc);
      }
#pragma unroll
      for (int r = 0; r < 8; ++r) {                   // causal + expert-active mask
        int mk = ms0 + halfid * 8 + r;
        bool valid = actq && (rw[(size_t)e * BN + b * Nq + mk] > 0.f) &&
                     (pids[b * Nq + mk] <= posq);
        sacc[r] = valid ? sacc[r] * scale : -1.0e9f;
      }
      st[s] = sacc;
    }
    // online softmax over keys (rows of S^T): in-lane + one shfl_xor(16)
    float mloc = -3.0e38f;
#pragma unroll
    for (int s = 0; s < 2; ++s)
#pragma unroll
      for (int r = 0; r < 8; ++r) mloc = fmaxf(mloc, st[s][r]);
    mloc = fmaxf(mloc, __shfl_xor(mloc, 16, 32));
    float mnew = fmaxf(mrun, mloc);
    float corr = __expf(mrun - mnew);
    lrun *= corr;
#pragma unroll
    for (int t = 0; t < 8; ++t)
#pragma unroll
      for (int r = 0; r < 8; ++r) oacc[t][r] *= corr;
    float sloc = 0.f;
#pragma unroll
    for (int s = 0; s < 2; ++s)
#pragma unroll
      for (int r = 0; r < 8; ++r) {
        float p = __expf(st[s][r] - mnew);
        st[s][r] = p; sloc += p;
      }
    lrun += sloc + __shfl_xor(sloc, 16, 32);
    mrun = mnew;

    AB16 pf = bfrag(st[0], st[1], halfid);            // P^T -> B fragment (K=32 keys)
#pragma unroll
    for (int t = 0; t < 8; ++t) {                     // O^T += V^T * P^T
      int vrow = KTILE16 + (t * 16 + c) * VROW16;
      AB16 vf;                                        // ds_load_b128 x2
      vf.q[0] = shmem[buf][vrow + 2 * halfid];
      vf.q[1] = shmem[buf][vrow + 2 * halfid + 1];
      oacc[t] = wmma_bf16(vf, pf, oacc[t]);
    }
  }

  // fold routing weight / softmax denominator into O^T columns (per query q=c)
  float wq = actq ? (rwq / lrun) : 0.f;
#pragma unroll
  for (int t = 0; t < 8; ++t)
#pragma unroll
    for (int r = 0; r < 8; ++r) oacc[t][r] *= wq;

  AB16 of[4];                                         // O^T -> B fragments (K=h)
#pragma unroll
  for (int ch = 0; ch < 4; ++ch) of[ch] = bfrag(oacc[2 * ch], oacc[2 * ch + 1], halfid);

  // fused output projection: y^T = w_o^T * O^T, accumulate into final output
  for (int dt = 0; dt < Dq / 16; ++dt) {
    v8f y;
#pragma unroll
    for (int r = 0; r < 8; ++r) y[r] = 0.f;
#pragma unroll
    for (int ch = 0; ch < 4; ++ch) {
      const __bf16* wrow = wotd + ((size_t)e * Dq + dt * 16 + c) * Hq + ch * 32;
      AB16 wf;
      wf.q[0] = ld16(wrow + halfid * 16);
      wf.q[1] = ld16(wrow + halfid * 16 + 8);
      y = wmma_bf16(wf, of[ch], y);
    }
    float* dstp = out + (size_t)(b * Nq + n0 + c) * Dq + dt * 16 + halfid * 8;
#pragma unroll
    for (int r = 0; r < 8; ++r) atomicAdd(dstp + r, y[r]);
  }
}

// ---------------------------------------------------------------- aux losses
__global__ void k_loss(const float* __restrict__ counts,
                       const float* __restrict__ probsum, float* __restrict__ out) {
  if (threadIdx.x == 0 && blockIdx.x == 0) {
    float s = 0.f, mx = 0.f;
    for (int e = 0; e < Eq; ++e) {
      s += (counts[e] / (float)(BN * Kq)) * (probsum[e] / (float)BN);
      mx = fmaxf(mx, counts[e]);
    }
    out[0] = (float)Eq * s;
    out[1] = mx / ((float)(BN * Kq) / (float)Eq) - 1.f;
  }
}

// ---------------------------------------------------------------- launch
extern "C" void kernel_launch(void* const* d_in, const int* in_sizes, int n_in,
                              void* d_out, int out_size, void* d_ws, size_t ws_size,
                              hipStream_t stream) {
  (void)in_sizes; (void)n_in; (void)ws_size;
  const float* x    = (const float*)d_in[0];
  const int*   pids = (const int*)d_in[1];
  const float* wr   = (const float*)d_in[2];
  const float* wqs  = (const float*)d_in[3];
  const float* wks  = (const float*)d_in[4];
  const float* wvs  = (const float*)d_in[5];
  const float* wos  = (const float*)d_in[6];
  float* out = (float*)d_out;

  char* ws = (char*)d_ws;
  size_t off = 0;
  auto alloc = [&](size_t bytes) -> void* {
    void* p = ws + off;
    off = (off + bytes + 255) & ~(size_t)255;
    return p;
  };
  __bf16* xbf  = (__bf16*)alloc((size_t)BN * Dq * 2);       // 16 MB
  __bf16* wtq  = (__bf16*)alloc((size_t)Eq * Hq * Dq * 2);  // 2 MB each
  __bf16* wtk  = (__bf16*)alloc((size_t)Eq * Hq * Dq * 2);
  __bf16* wtv  = (__bf16*)alloc((size_t)Eq * Hq * Dq * 2);
  __bf16* wotd = (__bf16*)alloc((size_t)Eq * Dq * Hq * 2);
  __bf16* qbf  = (__bf16*)alloc((size_t)Eq * BN * Hq * 2);  // 16.8 MB each
  __bf16* kbf  = (__bf16*)alloc((size_t)Eq * BN * Hq * 2);
  __bf16* vt   = (__bf16*)alloc((size_t)Eq * BN * Hq * 2);
  float*  rw   = (float*)alloc((size_t)Eq * BN * 4);
  float*  cnts = (float*)alloc(64 * 4);                     // counts[E] + probsum[E]
  float*  psum = cnts + Eq;

  hipMemsetAsync(rw, 0, (size_t)Eq * BN * 4, stream);
  hipMemsetAsync(cnts, 0, 64 * 4, stream);
  hipMemsetAsync(d_out, 0, (size_t)out_size * 4, stream);

  // converts / transposes
  k_cvt<<<(BN * Dq + 255) / 256, 256, 0, stream>>>(x, xbf, BN * Dq);
  int wn = Eq * Dq * Hq;
  k_tr<<<(wn + 255) / 256, 256, 0, stream>>>(wqs, wtq, Dq, Hq);  // (E,D,H)->(E,H,D)
  k_tr<<<(wn + 255) / 256, 256, 0, stream>>>(wks, wtk, Dq, Hq);
  k_tr<<<(wn + 255) / 256, 256, 0, stream>>>(wvs, wtv, Dq, Hq);
  k_tr<<<(wn + 255) / 256, 256, 0, stream>>>(wos, wotd, Hq, Dq); // (E,H,D)->(E,D,H)

  // router
  k_router<<<BN, 32, 0, stream>>>(x, wr, rw, cnts, psum);

  // QKV projections (WMMA) + RoPE
  k_qkv<<<dim3(BN / 16, Eq, 3), 32, 0, stream>>>(xbf, wtq, wtk, wtv, pids,
                                                 qbf, kbf, vt);

  // flash attention (TDM-staged K/V) + fused output projection (WMMA)
  k_attn<<<dim3(Nq / 16, Bq, Eq), 32, 0, stream>>>(qbf, kbf, vt, wotd, rw, pids, out);

  // scalars
  k_loss<<<1, 32, 0, stream>>>(cnts, psum, out + (size_t)BN * Dq);
}
